// BiT_Phoneme_81174881894828
// MI455X (gfx1250) — matrix operand order repaired and checked by reading, not compiled, not run on hardware
//
#include <hip/hip_runtime.h>
#include <hip/hip_bf16.h>

typedef __attribute__((ext_vector_type(16))) _Float16 v16h;
typedef __attribute__((ext_vector_type(8)))  _Float16 v8h;
typedef __attribute__((ext_vector_type(8)))  float    v8f;

union V16H { v16h v; _Float16 h[16]; v8h q[2]; };
union V8F  { v8f  v; float    f[8];  };

#define WMMA_F32_16x16x32_F16(A, B, C) \
    __builtin_amdgcn_wmma_f32_16x16x32_f16(false, (A), false, (B), (short)0, (C), false, false)

__device__ __forceinline__ void pack8(V16H& dst, int o, float4 a, float4 b) {
    dst.h[o + 0] = (_Float16)a.x; dst.h[o + 1] = (_Float16)a.y;
    dst.h[o + 2] = (_Float16)a.z; dst.h[o + 3] = (_Float16)a.w;
    dst.h[o + 4] = (_Float16)b.x; dst.h[o + 5] = (_Float16)b.y;
    dst.h[o + 6] = (_Float16)b.z; dst.h[o + 7] = (_Float16)b.w;
}

// ---------------------------------------------------------------------------
// Weight prep: fp32 row-major [mat][K][N] -> f16 transposed [mat][N][K].
// Coalesced reads; run once per launch before the GEMMs.
// ---------------------------------------------------------------------------
__global__ void convT_kernel(const float* __restrict__ src, _Float16* __restrict__ dst,
                             int K, int N, long total) {
    long i = (long)blockIdx.x * blockDim.x + threadIdx.x;
    if (i >= total) return;
    long kn  = (long)K * N;
    long mat = i / kn;
    long r   = i - mat * kn;
    int  n   = (int)(r % N);
    int  k   = (int)(r / N);
    dst[mat * kn + (size_t)n * K + k] = (_Float16)src[i];
}

// V staging for attention: vT[b][h][d][s] = (f16) qkv[b][s][1024 + h*64 + d]
__global__ void vT_kernel(const float* __restrict__ qkv, _Float16* __restrict__ vT) {
    long i = (long)blockIdx.x * blockDim.x + threadIdx.x;   // over B*S*512
    if (i >= (long)16 * 512 * 512) return;
    int c = (int)(i % 512);
    int s = (int)((i / 512) % 512);
    int b = (int)(i / (512 * 512));
    int h = c >> 6, d = c & 63;
    vT[(((size_t)b * 8 + h) * 64 + d) * 512 + s] =
        (_Float16)qkv[((size_t)b * 512 + s) * 1536 + 1024 + c];
}

// ---------------------------------------------------------------------------
// Gaussian depthwise smoothing over time (K=20, sigma=2, zero pad 'same').
// ---------------------------------------------------------------------------
__global__ void smooth_kernel(const float* __restrict__ x, float* __restrict__ y,
                              int T, int F, long total) {
    long i = (long)blockIdx.x * blockDim.x + threadIdx.x;
    if (i >= total) return;
    int  f  = (int)(i % F);
    long bt = i / F;
    int  t  = (int)(bt % T);
    long b  = bt / T;
    float acc = 0.f, wsum = 0.f;
#pragma unroll
    for (int k = 0; k < 20; ++k) {
        float d = ((float)k - 9.5f) * 0.5f;
        float w = __expf(-0.5f * d * d);
        wsum += w;
        int tt = t - 9 + k;
        if (tt >= 0 && tt < T)
            acc += w * x[(b * T + tt) * (long)F + f];
    }
    y[i] = acc / wsum;
}

// ---------------------------------------------------------------------------
// LayerNorm over last dim C (one block per row).
// ---------------------------------------------------------------------------
__global__ void ln_kernel(const float* __restrict__ X, const float* __restrict__ g,
                          const float* __restrict__ bta, float* __restrict__ Y, int C) {
    __shared__ float ssum[256];
    __shared__ float ssq[256];
    int row = blockIdx.x;
    int tid = threadIdx.x;
    const float* xr = X + (size_t)row * C;
    float* yr = Y + (size_t)row * C;
    float s = 0.f, q = 0.f;
    for (int i = tid; i < C; i += blockDim.x) { float v = xr[i]; s += v; q += v * v; }
    ssum[tid] = s; ssq[tid] = q;
    __syncthreads();
    for (int off = 128; off > 0; off >>= 1) {
        if (tid < off) { ssum[tid] += ssum[tid + off]; ssq[tid] += ssq[tid + off]; }
        __syncthreads();
    }
    float mean = ssum[0] / (float)C;
    float var  = ssq[0] / (float)C - mean * mean;
    float inv  = rsqrtf(var + 1e-5f);
    for (int i = tid; i < C; i += blockDim.x)
        yr[i] = (xr[i] - mean) * inv * g[i] + bta[i];
}

// ---------------------------------------------------------------------------
// WMMA GEMM: C[M,N] = act(A[M,K] @ W[K,N] + bias + resid), W pre-transposed
// to f16 [N][K]. One wave per 16x64 C strip (A fragment reused 4x).
// B fragment = 2x global_load_b128 of contiguous f16 per tile.
// act: 0=none, 1=softsign, 2=exact GELU. Optional per-batch weights (dayIdx).
// ---------------------------------------------------------------------------
__global__ __launch_bounds__(32)
void wmma_gemm_kernel(const float* __restrict__ A, const _Float16* __restrict__ Wt,
                      const float* __restrict__ bias, const float* __restrict__ resid,
                      float* __restrict__ C,
                      int M, int N, int K, int lda, int ldc, int act,
                      const int* __restrict__ dayIdx, int rowsPerBatch,
                      long wBatchStride, int biasBatchStride) {
    int lane = threadIdx.x;
    int half = lane >> 4;       // 0: lanes 0-15, 1: lanes 16-31
    int nn   = lane & 15;
    int m0 = blockIdx.y * 16;
    int n0 = blockIdx.x * 64;

    const _Float16* Wp = Wt;
    const float* bp = bias;
    if (dayIdx) {
        int bidx = dayIdx[m0 / rowsPerBatch];
        Wp += (size_t)bidx * wBatchStride;
        if (bp) bp += (size_t)bidx * biasBatchStride;
    }

    int arow = m0 + nn;
    if (arow >= M) arow = M - 1;
    const float* Ap = A + (size_t)arow * lda;

    int col[4];
    const _Float16* wrow[4];
#pragma unroll
    for (int j = 0; j < 4; ++j) {
        col[j] = n0 + j * 16 + nn;
        int cl = (col[j] < N) ? col[j] : (N - 1);     // clamp: branch-free k-loop
        wrow[j] = Wp + (size_t)cl * K + half * 16;
    }

    v8f acc[4] = {{}, {}, {}, {}};
    for (int k0 = 0; k0 < K; k0 += 32) {
        // A fragment: e -> K = k0 + 16*half + e (contiguous)
        V16H a;
        const float4* pa0 = (const float4*)(Ap + k0 + half * 16);
        const float4* pa1 = (const float4*)(Ap + k0 + 8 + half * 16);
        pack8(a, 0, pa0[0], pa0[1]);
        pack8(a, 8, pa1[0], pa1[1]);
#pragma unroll
        for (int j = 0; j < 4; ++j) {
            // B fragment: e -> K = k0 + 16*half + e (16 contiguous f16)
            const v8h* pb = (const v8h*)(wrow[j] + k0);
            V16H b;
            b.q[0] = pb[0];
            b.q[1] = pb[1];
            acc[j] = WMMA_F32_16x16x32_F16(a.v, b.v, acc[j]);
        }
    }

#pragma unroll
    for (int j = 0; j < 4; ++j) {
        float bv = (bp && col[j] < N) ? bp[col[j]] : 0.f;
        V8F out; out.v = acc[j];
#pragma unroll
        for (int r = 0; r < 8; ++r) {
            int row = m0 + half * 8 + r;      // C layout: VGPR r -> M = r + 8*half
            if (row < M && col[j] < N) {
                float v = out.f[r] + bv;
                if (resid) v += resid[(size_t)row * ldc + col[j]];
                if (act == 1)      v = v / (1.0f + fabsf(v));                      // softsign
                else if (act == 2) v = 0.5f * v * (1.0f + erff(v * 0.70710678f));  // exact gelu
                C[(size_t)row * ldc + col[j]] = v;
            }
        }
    }
}

// ---------------------------------------------------------------------------
// Fused attention: one wave per (b, head, 16-row q tile). S=512, DHEAD=64.
// Q/K read fp32 from qkv; V read from pre-staged f16 vT[b][h][d][s].
// scores (16 x 512) staged in LDS; softmax in LDS; P @ V via WMMA.
// ---------------------------------------------------------------------------
#define ATT_S 512
__global__ __launch_bounds__(32)
void attn_kernel(const float* __restrict__ qkv, const _Float16* __restrict__ vT,
                 const float* __restrict__ relT, float* __restrict__ o) {
    __shared__ float sc[16][ATT_S];
    __shared__ float red[2][16];
    int lane = threadIdx.x;
    int half = lane >> 4;
    int nn   = lane & 15;
    int qt = blockIdx.x, h = blockIdx.y, b = blockIdx.z;
    const float scale = 0.125f;  // 64^-0.5
    size_t base = (size_t)b * ATT_S * 1536;

    // Q tile (16x64) as two 16x32 f16 A operands
    V16H aq0, aq1;
    {
        const float* qrow = qkv + base + (size_t)(qt * 16 + nn) * 1536 + h * 64;
        const float4* p0 = (const float4*)(qrow + half * 16);
        const float4* p1 = (const float4*)(qrow + 8 + half * 16);
        const float4* p2 = (const float4*)(qrow + 32 + half * 16);
        const float4* p3 = (const float4*)(qrow + 40 + half * 16);
        pack8(aq0, 0, p0[0], p0[1]);
        pack8(aq0, 8, p1[0], p1[1]);
        pack8(aq1, 0, p2[0], p2[1]);
        pack8(aq1, 8, p3[0], p3[1]);
    }

    // scores = Q K^T * scale + rel_bias, causal mask -> LDS
    for (int kt = 0; kt < 32; ++kt) {
        v8f acc = {};
        const float* krow = qkv + base + (size_t)(kt * 16 + nn) * 1536 + 512 + h * 64;
        V16H b0, b1;   // B fragment: d = 16*half + e -> contiguous
        const float4* pk0 = (const float4*)(krow + half * 16);
        const float4* pk1 = (const float4*)(krow + 32 + half * 16);
        pack8(b0, 0, pk0[0], pk0[1]);
        pack8(b0, 8, pk0[2], pk0[3]);
        pack8(b1, 0, pk1[0], pk1[1]);
        pack8(b1, 8, pk1[2], pk1[3]);
        acc = WMMA_F32_16x16x32_F16(aq0.v, b0.v, acc);
        acc = WMMA_F32_16x16x32_F16(aq1.v, b1.v, acc);
        int ki = kt * 16 + nn;
        V8F out; out.v = acc;
#pragma unroll
        for (int r = 0; r < 8; ++r) {
            int qr = half * 8 + r;
            int qi = qt * 16 + qr;
            int dlt = qi - ki;
            if (dlt >  199) dlt =  199;
            if (dlt < -199) dlt = -199;
            float s = out.f[r] * scale + relT[dlt + 199];
            if (ki > qi) s = -1e9f;                // causal mask
            sc[qr][ki] = s;
        }
    }
    __syncthreads();

    // softmax: row = nn, each half covers 256 cols; merge halves via LDS
    {
        float* rowp = &sc[nn][half * 256];
        float mx = -3.4e38f;
        for (int j = 0; j < 256; ++j) mx = fmaxf(mx, rowp[j]);
        red[half][nn] = mx;
        __syncthreads();
        mx = fmaxf(red[0][nn], red[1][nn]);
        __syncthreads();
        float sum = 0.f;
        for (int j = 0; j < 256; ++j) {
            float e = __expf(rowp[j] - mx);
            rowp[j] = e;
            sum += e;
        }
        red[half][nn] = sum;
        __syncthreads();
        float inv = 1.0f / (red[0][nn] + red[1][nn]);
        for (int j = 0; j < 256; ++j) rowp[j] *= inv;
    }
    __syncthreads();

    // O(16x64) = P(16x512) @ V(512x64); V from f16 vT (contiguous in s)
#pragma unroll
    for (int nt = 0; nt < 4; ++nt) {
        v8f acc = {};
        const _Float16* vrow = vT + (((size_t)b * 8 + h) * 64 + nt * 16 + nn) * ATT_S + half * 16;
        for (int kc = 0; kc < 16; ++kc) {
            V16H ap, bv;
            const float4* pp0 = (const float4*)(&sc[nn][kc * 32 + half * 16]);
            const float4* pp1 = (const float4*)(&sc[nn][kc * 32 + 8 + half * 16]);
            pack8(ap, 0, pp0[0], pp0[1]);
            pack8(ap, 8, pp1[0], pp1[1]);
            const v8h* pv = (const v8h*)(vrow + kc * 32);
            bv.q[0] = pv[0];
            bv.q[1] = pv[1];
            acc = WMMA_F32_16x16x32_F16(ap.v, bv.v, acc);
        }
        V8F out; out.v = acc;
#pragma unroll
        for (int r = 0; r < 8; ++r) {
            int row = qt * 16 + half * 8 + r;
            o[((size_t)b * ATT_S + row) * 512 + h * 64 + nt * 16 + nn] = out.f[r];
        }
    }
}

// ---------------------------------------------------------------------------
// Host orchestration
// ---------------------------------------------------------------------------
extern "C" void kernel_launch(void* const* d_in, const int* in_sizes, int n_in,
                              void* d_out, int out_size, void* d_ws, size_t ws_size,
                              hipStream_t stream) {
    const float* neural = (const float*)d_in[0];
    // d_in[1] = X_len: eval path, unused
    const int*   dayIdx = (const int*)d_in[2];
    const float* dayW   = (const float*)d_in[3];   // (24,256,256)
    const float* dayB   = (const float*)d_in[4];   // (24,1,256)
    const float* pLNg   = (const float*)d_in[5];
    const float* pLNb   = (const float*)d_in[6];
    const float* patchW = (const float*)d_in[7];   // (1024,512)
    const float* patchB = (const float*)d_in[8];
    const float* eLNg   = (const float*)d_in[9];
    const float* eLNb   = (const float*)d_in[10];
    const float* fLNg   = (const float*)d_in[11];
    const float* fLNb   = (const float*)d_in[12];
    const float* projW  = (const float*)d_in[13];  // (512,41)
    const float* projB  = (const float*)d_in[14];
    const float* aLNg   = (const float*)d_in[15];  // (6,512)
    const float* aLNb   = (const float*)d_in[16];
    const float* wqkv   = (const float*)d_in[17];  // (6,512,1536)
    const float* wo     = (const float*)d_in[18];  // (6,512,512)
    const float* bo     = (const float*)d_in[19];  // (6,512)
    const float* relT   = (const float*)d_in[20];  // (6,399,1)
    const float* ffLNg  = (const float*)d_in[21];
    const float* ffLNb  = (const float*)d_in[22];
    const float* w1     = (const float*)d_in[23];  // (6,512,2048)
    const float* b1     = (const float*)d_in[24];  // (6,2048)
    const float* w2     = (const float*)d_in[25];  // (6,2048,512)
    const float* b2     = (const float*)d_in[26];  // (6,512)

    const int Bn = 16, T = 2048, F = 256, S = 512, DIM = 512, HEADS = 8, MLP = 2048;
    const int NCLS = 41;
    const int BS = Bn * S;   // 8192
    const int BT = Bn * T;   // 32768

    // fp32 arena
    float* ws   = (float*)d_ws;
    float* buf0 = ws;                    //  8,388,608 floats
    float* buf1 = ws + 8388608;          //  8,388,608
    float* x    = ws + 16777216;         //  4,194,304
    float* xn   = ws + 20971520;         //  4,194,304
    float* qkv  = ws + 25165824;         // 12,582,912
    float* o    = ws + 37748736;         //  4,194,304
    float* ff   = buf0;                  // 16,777,216 (spans buf0+buf1)

    // f16 arena (transposed weights + staged V), 16B-aligned offsets
    _Float16* h16     = (_Float16*)(ws + 41943040);
    _Float16* dayWT   = h16;                 // 24*256*256  = 1,572,864
    _Float16* patchWT = h16 + 1572864;       // 512*1024    =   524,288
    _Float16* qkvWT   = h16 + 2097152;       // 6*1536*512  = 4,718,592
    _Float16* woWT    = h16 + 6815744;       // 6*512*512   = 1,572,864
    _Float16* w1T     = h16 + 8388608;       // 6*2048*512  = 6,291,456
    _Float16* w2T     = h16 + 14680064;      // 6*512*2048  = 6,291,456
    _Float16* projWT  = h16 + 20971520;      // 41*512      =    20,992
    _Float16* vT      = h16 + 20992512;      // 16*8*64*512 = 4,194,304

    dim3 wave(32);
    auto cdiv = [](long a, long b) { return (unsigned)((a + b - 1) / b); };

    // 0) weight prep: fp32 [K][N] -> f16 [N][K]
    convT_kernel<<<cdiv(24L * 256 * 256, 256), 256, 0, stream>>>(dayW,   dayWT,   256,  256,  24L * 256 * 256);
    convT_kernel<<<cdiv(1024L * 512, 256),     256, 0, stream>>>(patchW, patchWT, 1024, 512,  1024L * 512);
    convT_kernel<<<cdiv(6L * 512 * 1536, 256), 256, 0, stream>>>(wqkv,   qkvWT,   512,  1536, 6L * 512 * 1536);
    convT_kernel<<<cdiv(6L * 512 * 512, 256),  256, 0, stream>>>(wo,     woWT,    512,  512,  6L * 512 * 512);
    convT_kernel<<<cdiv(6L * 512 * 2048, 256), 256, 0, stream>>>(w1,     w1T,     512,  2048, 6L * 512 * 2048);
    convT_kernel<<<cdiv(6L * 2048 * 512, 256), 256, 0, stream>>>(w2,     w2T,     2048, 512,  6L * 2048 * 512);
    convT_kernel<<<cdiv(512L * 41, 256),       256, 0, stream>>>(projW,  projWT,  512,  41,   512L * 41);

    // 1) Gaussian smoothing: neural -> buf0
    {
        long total = (long)Bn * T * F;
        smooth_kernel<<<cdiv(total, 256), 256, 0, stream>>>(neural, buf0, T, F, total);
    }
    // 2) per-day affine + softsign: buf0 @ dayW[dayIdx] + dayB -> buf1
    wmma_gemm_kernel<<<dim3(F / 64, BT / 16), wave, 0, stream>>>(
        buf0, dayWT, dayB, nullptr, buf1, BT, F, F, F, F, /*act=*/1,
        dayIdx, /*rowsPerBatch=*/T, (long)F * F, F);
    // 3) patch LN over 1024 (contiguous reshape): buf1 -> buf0
    ln_kernel<<<BS, 256, 0, stream>>>(buf1, pLNg, pLNb, buf0, 1024);
    // 4) patch embed -> xn; embedding LN -> x
    wmma_gemm_kernel<<<dim3(DIM / 64, BS / 16), wave, 0, stream>>>(
        buf0, patchWT, patchB, nullptr, xn, BS, DIM, 1024, 1024, DIM, 0,
        nullptr, 0, 0, 0);
    ln_kernel<<<BS, 256, 0, stream>>>(xn, eLNg, eLNb, x, DIM);

    for (int l = 0; l < 6; ++l) {
        // attention
        ln_kernel<<<BS, 256, 0, stream>>>(x, aLNg + l * DIM, aLNb + l * DIM, xn, DIM);
        wmma_gemm_kernel<<<dim3(1536 / 64, BS / 16), wave, 0, stream>>>(
            xn, qkvWT + (size_t)l * 1536 * DIM, nullptr, nullptr, qkv,
            BS, 1536, DIM, DIM, 1536, 0, nullptr, 0, 0, 0);
        vT_kernel<<<cdiv((long)Bn * S * 512, 256), 256, 0, stream>>>(qkv, vT);
        attn_kernel<<<dim3(S / 16, HEADS, Bn), wave, 0, stream>>>(qkv, vT, relT + l * 399, o);
        wmma_gemm_kernel<<<dim3(DIM / 64, BS / 16), wave, 0, stream>>>(
            o, woWT + (size_t)l * DIM * DIM, bo + l * DIM, /*resid=*/x, x,
            BS, DIM, DIM, DIM, DIM, 0, nullptr, 0, 0, 0);
        // FFN
        ln_kernel<<<BS, 256, 0, stream>>>(x, ffLNg + l * DIM, ffLNb + l * DIM, xn, DIM);
        wmma_gemm_kernel<<<dim3(MLP / 64, BS / 16), wave, 0, stream>>>(
            xn, w1T + (size_t)l * MLP * DIM, b1 + l * MLP, nullptr, ff,
            BS, MLP, DIM, DIM, MLP, /*act=*/2, nullptr, 0, 0, 0);
        wmma_gemm_kernel<<<dim3(DIM / 64, BS / 16), wave, 0, stream>>>(
            ff, w2T + (size_t)l * DIM * MLP, b2 + l * DIM, /*resid=*/x, x,
            BS, DIM, MLP, MLP, DIM, 0, nullptr, 0, 0, 0);
    }

    // final LN + classifier
    ln_kernel<<<BS, 256, 0, stream>>>(x, fLNg, fLNb, xn, DIM);
    wmma_gemm_kernel<<<dim3((NCLS + 63) / 64, BS / 16), wave, 0, stream>>>(
        xn, projWT, projB, nullptr, (float*)d_out,
        BS, NCLS, DIM, DIM, NCLS, 0, nullptr, 0, 0, 0);
}
